// RelationalTransformer_9234179687213
// MI455X (gfx1250) — matrix operand order repaired and checked by reading, not compiled, not run on hardware
//
#include <hip/hip_runtime.h>

typedef _Float16 f16;
typedef __attribute__((ext_vector_type(16))) _Float16 v16h;
typedef __attribute__((ext_vector_type(8)))  _Float16 h8;
typedef __attribute__((ext_vector_type(8)))  float    v8f;
typedef __attribute__((ext_vector_type(4)))  unsigned int u32x4;
typedef __attribute__((ext_vector_type(8)))  int i32x8;
typedef __attribute__((ext_vector_type(4)))  int i32x4;

#define HAVE_TDM __has_builtin(__builtin_amdgcn_tensor_load_to_lds)

// ---------------------------------------------------------------------------
// WMMA fragment loaders (layouts per CDNA5 ISA 7.12.2, wave32)
// A (16x32 f16, row-major, lda in halves):
//   lanes 0-15 : row m=lane,   halves K=[kk..kk+7] and [kk+16..kk+23]
//   lanes 16-31: row m=lane-16,halves K=[kk+8..kk+15] and [kk+24..kk+31]
// B (32x16 f16) built from W row-major (row n = output col n, ldw in halves):
//   lanes 0-15 : col n=lane,    halves K=[kk..kk+7] and [kk+16..kk+23]
//   lanes 16-31: col n=lane-16, halves K=[kk+8..kk+15] and [kk+24..kk+31]
// ---------------------------------------------------------------------------
__device__ __forceinline__ v16h load_a_frag(const f16* __restrict__ A, int lda,
                                            int kk, int lane) {
  int r  = lane & 15;
  int hi = lane >> 4;
  const f16* base = A + (size_t)r * lda + kk + hi * 8;
  h8 lo = *(const h8*)(base);
  h8 hv = *(const h8*)(base + 16);
  v16h a;
#pragma unroll
  for (int i = 0; i < 8; ++i) { a[i] = lo[i]; a[i + 8] = hv[i]; }
  return a;
}

__device__ __forceinline__ v16h load_b_frag(const f16* __restrict__ W, int ldw,
                                            int kk, int lane) {
  int n  = lane & 15;
  int hi = lane >> 4;
  const f16* base = W + (size_t)n * ldw + kk + hi * 8;
  h8 lo = *(const h8*)(base);
  h8 hv = *(const h8*)(base + 16);
  v16h b;
#pragma unroll
  for (int i = 0; i < 8; ++i) { b[i] = lo[i]; b[i + 8] = hv[i]; }
  return b;
}

// ---------------------------------------------------------------------------
// TDM: issue a 2D tile load (tile_rows x tile_k f16, row stride in elements)
// into LDS at byte offset lds_off. D# packing per CDNA5 ISA 8.3/8.4.
// ---------------------------------------------------------------------------
#if HAVE_TDM
__device__ __forceinline__ void tdm_load_tile(unsigned long long gaddr,
                                              unsigned lds_off,
                                              unsigned tile_k, unsigned tile_rows,
                                              unsigned long long row_stride_elems) {
  u32x4 g0;
  g0[0] = 1u;                                           // count=1, user mode
  g0[1] = lds_off;                                      // lds_addr (bytes)
  g0[2] = (unsigned)(gaddr & 0xFFFFFFFFull);            // global_addr[31:0]
  g0[3] = (unsigned)((gaddr >> 32) & 0x01FFFFFFull)     // global_addr[56:32]
          | (2u << 30);                                 // type=2 ("image")
  i32x8 g1;
  g1[0] = (int)(1u << 16);                              // wg_mask=0, data_size=1 (2B)
  g1[1] = (int)((tile_k & 0xFFFFu) << 16);              // tensor_dim0[15:0] (=tile)
  g1[2] = (int)((tile_rows & 0xFFFFu) << 16);           // dim0[31:16]=0, tensor_dim1[15:0]
  g1[3] = (int)((tile_k & 0xFFFFu) << 16);              // dim1[31:16]=0, tile_dim0
  g1[4] = (int)(tile_rows & 0xFFFFu);                   // tile_dim1, tile_dim2=0
  g1[5] = (int)(row_stride_elems & 0xFFFFFFFFull);      // tensor_dim0_stride[31:0]
  g1[6] = (int)((row_stride_elems >> 32) & 0xFFFFull);  // stride[47:32], dim1_stride=0
  g1[7] = 0;
  i32x4 z4 = {};
#if __clang_major__ >= 23
  i32x8 z8 = {};
  __builtin_amdgcn_tensor_load_to_lds(g0, g1, z4, z4, z8, 0);
#else
  __builtin_amdgcn_tensor_load_to_lds(g0, g1, z4, z4, 0);
#endif
}
#endif

__device__ __forceinline__ unsigned lds_byte_off(const f16* p) {
  // addrspace(3) pointers hold the LDS byte offset in the low 32 bits
  return (unsigned)(unsigned long long)(uintptr_t)p;
}

// fallback staging: 128x64 f16 tile, cooperative synchronous copy
__device__ __forceinline__ void stage_tile(f16* dst, const f16* __restrict__ src,
                                           int row_stride, int tid) {
  for (int i = tid; i < 1024; i += 256) {  // 1024 chunks of 8 halves
    const int row = i >> 3, col = (i & 7) * 8;
    *(h8*)(dst + row * 64 + col) = *(const h8*)(src + (size_t)row * row_stride + col);
  }
}

// ---------------------------------------------------------------------------
// LDS-tiled WMMA GEMM for the big K/V projections.
// C[m,n] = sum_k A[m,k]*W[n,k] + bias[n], K = 1024 fixed, lda = ldw = ldc = 1024.
// Block tile 128x128, 8 waves (wave w: rows w*16..w*16+15 x 128 cols),
// K-chunks of 64, double-buffered LDS filled by the Tensor Data Mover.
// Inner loop: batch all 8 B-fragment LDS loads first (one dscnt wait), then
// fire the 8 independent WMMAs back-to-back.
// grid = (N/128, M/128), block = 256.
// ---------------------------------------------------------------------------
__global__ void __launch_bounds__(256) gemm_tdm_wmma(
    const f16* __restrict__ A, const f16* __restrict__ W,
    const float* __restrict__ bias, f16* __restrict__ outF16) {
  __shared__ f16 Ab[2][128][64];
  __shared__ f16 Bb[2][128][64];
  const int tid = threadIdx.x;
  const int lane = tid & 31, wave = tid >> 5;
  const int m_base = blockIdx.y * 128;
  const int n_base = blockIdx.x * 128;
  const f16* Ag = A + (size_t)m_base * 1024;
  const f16* Wg = W + (size_t)n_base * 1024;

  v8f acc[8];
#pragma unroll
  for (int i = 0; i < 8; ++i) acc[i] = (v8f){};

  constexpr int STEPS = 16;  // K=1024 / KC=64

#if HAVE_TDM
  if (wave == 0) {
    tdm_load_tile((unsigned long long)(uintptr_t)Ag, lds_byte_off(&Ab[0][0][0]), 64, 128, 1024);
    tdm_load_tile((unsigned long long)(uintptr_t)Wg, lds_byte_off(&Bb[0][0][0]), 64, 128, 1024);
  }
#else
  stage_tile(&Ab[0][0][0], Ag, 1024, tid);
  stage_tile(&Bb[0][0][0], Wg, 1024, tid);
#endif

  for (int s = 0; s < STEPS; ++s) {
    const int p = s & 1;
    if (s + 1 < STEPS) {
      const int kc = (s + 1) * 64;
#if HAVE_TDM
      if (wave == 0) {
        tdm_load_tile((unsigned long long)(uintptr_t)(Ag + kc),
                      lds_byte_off(&Ab[p ^ 1][0][0]), 64, 128, 1024);
        tdm_load_tile((unsigned long long)(uintptr_t)(Wg + kc),
                      lds_byte_off(&Bb[p ^ 1][0][0]), 64, 128, 1024);
      }
#else
      stage_tile(&Ab[p ^ 1][0][0], Ag + kc, 1024, tid);
      stage_tile(&Bb[p ^ 1][0][0], Wg + kc, 1024, tid);
#endif
    }
#if HAVE_TDM
    // per-wave in-order completion: <=2 outstanding means the older pair landed
    if (s + 1 < STEPS) __builtin_amdgcn_s_wait_tensorcnt(2);
    else               __builtin_amdgcn_s_wait_tensorcnt(0);
#endif
    __syncthreads();
#pragma unroll
    for (int kk = 0; kk < 64; kk += 32) {
      v16h a = load_a_frag(&Ab[p][wave * 16][0], 64, kk, lane);
      v16h bf[8];
#pragma unroll
      for (int nt = 0; nt < 8; ++nt)          // batched LDS loads: one wait
        bf[nt] = load_b_frag(&Bb[p][nt * 16][0], 64, kk, lane);
#pragma unroll
      for (int nt = 0; nt < 8; ++nt)          // 8 independent WMMAs back-to-back
        acc[nt] = __builtin_amdgcn_wmma_f32_16x16x32_f16(false, a, false, bf[nt],
                                                         (short)0, acc[nt], false, false);
    }
    __syncthreads();
  }

  const int cc = lane & 15;
  const int hi = lane >> 4;
#pragma unroll
  for (int nt = 0; nt < 8; ++nt) {
    const float bv = bias ? bias[n_base + nt * 16 + cc] : 0.0f;
#pragma unroll
    for (int j = 0; j < 8; ++j) {
      const int m = m_base + wave * 16 + j + hi * 8;
      outF16[(size_t)m * 1024 + n_base + nt * 16 + cc] = (f16)(acc[nt][j] + bv);
    }
  }
}

// ---------------------------------------------------------------------------
// Generic register-direct WMMA GEMM (small M): C[m,n] = sum_k A[m,k]*W[n,k]+bias[n]
// One wave per 16x16 tile. grid = (N/16, M/16), block = 32.
// ---------------------------------------------------------------------------
__global__ void __launch_bounds__(32) gemm16_wmma(
    const f16* __restrict__ A, int lda,
    const f16* __restrict__ W, int ldw,
    const float* __restrict__ bias,
    float* __restrict__ outF32, f16* __restrict__ outF16, int ldc, int K) {
  const int lane = threadIdx.x;
  const int m0 = blockIdx.y * 16;
  const int n0 = blockIdx.x * 16;
  const f16* Ab = A + (size_t)m0 * lda;
  const f16* Wb = W + (size_t)n0 * ldw;
  v8f acc = {};
  for (int kk = 0; kk < K; kk += 32) {
    v16h af = load_a_frag(Ab, lda, kk, lane);
    v16h bf = load_b_frag(Wb, ldw, kk, lane);
    if (kk + 32 < K) {  // global_prefetch_b8 of next K-slice
      __builtin_prefetch(Ab + (size_t)(lane & 15) * lda + kk + 32, 0, 3);
      __builtin_prefetch(Wb + (size_t)(lane & 15) * ldw + kk + 32, 0, 3);
    }
    acc = __builtin_amdgcn_wmma_f32_16x16x32_f16(false, af, false, bf,
                                                 (short)0, acc, false, false);
  }
  const int cc = lane & 15;
  const int hi = lane >> 4;
  const float bv = bias ? bias[n0 + cc] : 0.0f;
#pragma unroll
  for (int j = 0; j < 8; ++j) {
    const int m = m0 + j + hi * 8;
    const float v = acc[j] + bv;
    if (outF32) outF32[(size_t)m * ldc + n0 + cc] = v;
    if (outF16) outF16[(size_t)m * ldc + n0 + cc] = (f16)v;
  }
}

// ---------------------------------------------------------------------------
// visual_feats[:, -1] -> f16
// ---------------------------------------------------------------------------
__global__ void cast_vf(const float* __restrict__ vf, f16* __restrict__ Xh) {
  const int idx = blockIdx.x * 256 + threadIdx.x;
  if (idx >= 32 * 1024 * 1024) return;
  const int b = idx >> 20;
  const int r = idx & 1048575;
  Xh[idx] = (f16)vf[((size_t)b * 4 + 3) * 1048576 + r];
}

__global__ void cvt_f32_f16(const float* __restrict__ s, f16* __restrict__ d, int n) {
  const int i = blockIdx.x * 256 + threadIdx.x;
  if (i < n) d[i] = (f16)s[i];
}

// ---------------------------------------------------------------------------
// query = emb[max(nodes,0)] + bboxes @ Wb^T + bb   -> f16 (480 x 1024)
// ---------------------------------------------------------------------------
__global__ void query_kernel(const int* __restrict__ nodes,
                             const float* __restrict__ bboxes,
                             const float* __restrict__ emb,
                             const float* __restrict__ Wb,
                             const float* __restrict__ bb,
                             f16* __restrict__ queryh) {
  const int idx = blockIdx.x * 256 + threadIdx.x;
  if (idx >= 480 * 1024) return;
  const int d = idx & 1023;
  const int bn = idx >> 10;
  int node = nodes[bn];
  int e = node > 0 ? node : 0;
  const float* bx = bboxes + (size_t)bn * 4;
  const float* w  = Wb + (size_t)d * 4;
  float v = emb[(size_t)e * 1024 + d] + bb[d] +
            bx[0] * w[0] + bx[1] * w[1] + bx[2] * w[2] + bx[3] * w[3];
  queryh[idx] = (f16)v;
}

// q rows (480x1024) -> padded (B,H,16,128), row n==15 zeroed
__global__ void repack_q(const f16* __restrict__ qrow, f16* __restrict__ qh) {
  const int idx = blockIdx.x * 256 + threadIdx.x;
  if (idx >= 32 * 8 * 16 * 128) return;
  const int d = idx & 127;
  int t = idx >> 7;
  const int n = t & 15; t >>= 4;
  const int h = t & 7;
  const int b = t >> 3;
  f16 v = (f16)0.0f;
  if (n < 15) v = qrow[((size_t)(b * 15 + n)) * 1024 + h * 128 + d];
  qh[idx] = v;
}

// ---------------------------------------------------------------------------
// Attention per (b,h): scores via WMMA (16x1024), softmax, ctx (VALU).
// grid = (H, B), block = 256 (8 waves). LDS = 64 KB score tile.
// ---------------------------------------------------------------------------
__global__ void __launch_bounds__(256) attn_kernel(
    const f16* __restrict__ qh,    // (B,H,16,128)
    const f16* __restrict__ Kh,    // (B*L, 1024) = (b,l,h,d)
    const f16* __restrict__ Vh,    // same layout
    f16* __restrict__ ctxh) {      // (B*15, 1024)
  __shared__ float sc[16][1024];
  const int h = blockIdx.x, b = blockIdx.y;
  const int tid = threadIdx.x;
  const int lane = tid & 31, wave = tid >> 5;
  const float scale = 0.08838834764831845f;  // 1/sqrt(128)

  const f16* Q = qh + ((size_t)(b * 8 + h)) * 16 * 128;
  v16h afr[4];
#pragma unroll
  for (int kk4 = 0; kk4 < 4; ++kk4) afr[kk4] = load_a_frag(Q, 128, kk4 * 32, lane);

  for (int lt = wave; lt < 64; lt += 8) {
    const f16* Kb = Kh + ((size_t)(b * 1024 + lt * 16)) * 1024 + h * 128;
    v8f acc = {};
#pragma unroll
    for (int kk4 = 0; kk4 < 4; ++kk4) {
      v16h bfr = load_b_frag(Kb, 1024, kk4 * 32, lane);
      acc = __builtin_amdgcn_wmma_f32_16x16x32_f16(false, afr[kk4], false, bfr,
                                                   (short)0, acc, false, false);
    }
    const int cc = lane & 15, hi = lane >> 4;
#pragma unroll
    for (int j = 0; j < 8; ++j) sc[j + hi * 8][lt * 16 + cc] = acc[j] * scale;
  }
  __syncthreads();

  // softmax over L: 16 threads per row, shuffle reduction within 16-lane group
  {
    const int m = tid >> 4, c0 = tid & 15;
    float lmax = -3.0e38f;
    for (int c = c0; c < 1024; c += 16) lmax = fmaxf(lmax, sc[m][c]);
    for (int off = 8; off >= 1; off >>= 1) lmax = fmaxf(lmax, __shfl_xor(lmax, off, 32));
    float lsum = 0.0f;
    for (int c = c0; c < 1024; c += 16) {
      float e = __expf(sc[m][c] - lmax);
      sc[m][c] = e;
      lsum += e;
    }
    for (int off = 8; off >= 1; off >>= 1) lsum += __shfl_xor(lsum, off, 32);
    const float inv = 1.0f / lsum;
    for (int c = c0; c < 1024; c += 16) sc[m][c] *= inv;
  }
  __syncthreads();

  // ctx = attn @ V  (15x128 outputs per block)
  for (int idx = tid; idx < 15 * 128; idx += 256) {
    const int m = idx >> 7, d = idx & 127;
    const f16* Vp = Vh + (size_t)b * 1024 * 1024 + h * 128 + d;
    float acc = 0.0f;
    for (int l = 0; l < 1024; ++l) acc += sc[m][l] * (float)Vp[(size_t)l * 1024];
    ctxh[((size_t)(b * 15 + m)) * 1024 + h * 128 + d] = (f16)acc;
  }
}

// ---------------------------------------------------------------------------
// edge_logits[b,i,j,c] = relu(h_i[b,i,:]+h_j[b,j,:]+b1) . W2[c,:] + b2[c]
// ---------------------------------------------------------------------------
__global__ void edge_kernel(const float* __restrict__ h_i,
                            const float* __restrict__ h_j,
                            const float* __restrict__ b1,
                            const float* __restrict__ W2,
                            const float* __restrict__ b2,
                            float* __restrict__ out) {
  const int idx = blockIdx.x * 256 + threadIdx.x;
  if (idx >= 32 * 15 * 15 * 18) return;
  const int c = idx % 18;
  int t = idx / 18;
  const int j = t % 15; t /= 15;
  const int i = t % 15;
  const int b = t / 15;
  const float* hp = h_i + (size_t)(b * 15 + i) * 512;
  const float* hq = h_j + (size_t)(b * 15 + j) * 512;
  const float* w  = W2 + (size_t)c * 512;
  float acc = b2[c];
  for (int k = 0; k < 512; ++k) {
    float p = hp[k] + hq[k] + b1[k];
    p = p > 0.0f ? p : 0.0f;
    acc += p * w[k];
  }
  out[idx] = acc;
}

// ---------------------------------------------------------------------------
// energy head: gfeat mean over valid nodes, 2-layer MLP. One block per b.
// ---------------------------------------------------------------------------
__global__ void __launch_bounds__(256) energy_kernel(
    const float* __restrict__ enriched, const int* __restrict__ nodes,
    const float* __restrict__ Wh1, const float* __restrict__ bh1,
    const float* __restrict__ Wh2, const float* __restrict__ bh2,
    float* __restrict__ out) {
  __shared__ float gf[1024];
  __shared__ float part[256];
  const int b = blockIdx.x;
  const int tid = threadIdx.x;
  int cnt = 0;
  for (int n = 0; n < 15; ++n)
    if (nodes[b * 15 + n] != -1) cnt++;
  const float inv = 1.0f / (float)(cnt > 0 ? cnt : 1);
  for (int d = tid; d < 1024; d += 256) {
    float s = 0.0f;
    for (int n = 0; n < 15; ++n)
      if (nodes[b * 15 + n] != -1) s += enriched[((size_t)(b * 15 + n)) * 1024 + d];
    gf[d] = s * inv;
  }
  __syncthreads();
  {
    const float* w = Wh1 + (size_t)tid * 1024;
    float acc = bh1[tid];
    for (int k = 0; k < 1024; ++k) acc += gf[k] * w[k];
    acc = acc > 0.0f ? acc : 0.0f;
    part[tid] = acc * Wh2[tid];
  }
  __syncthreads();
  if (tid == 0) {
    float e = bh2[0];
    for (int r = 0; r < 256; ++r) e += part[r];
    out[b] = e;
  }
}

// ---------------------------------------------------------------------------
// launch
// ---------------------------------------------------------------------------
extern "C" void kernel_launch(void* const* d_in, const int* in_sizes, int n_in,
                              void* d_out, int out_size, void* d_ws, size_t ws_size,
                              hipStream_t stream) {
  (void)in_sizes; (void)n_in; (void)out_size; (void)ws_size;
  const float* visual_feats = (const float*)d_in[0];
  const int*   nodes        = (const int*)  d_in[1];
  const float* bboxes       = (const float*)d_in[2];
  const float* emb          = (const float*)d_in[3];
  const float* Wb           = (const float*)d_in[4];
  const float* bb           = (const float*)d_in[5];
  const float* Wq           = (const float*)d_in[6];
  const float* bq           = (const float*)d_in[7];
  const float* Wk           = (const float*)d_in[8];
  const float* bk           = (const float*)d_in[9];
  const float* Wv           = (const float*)d_in[10];
  const float* bv           = (const float*)d_in[11];
  const float* Wo           = (const float*)d_in[12];
  const float* bo           = (const float*)d_in[13];
  const float* W1           = (const float*)d_in[14];
  const float* b1           = (const float*)d_in[15];
  const float* W2           = (const float*)d_in[16];
  const float* b2           = (const float*)d_in[17];
  const float* Wh1          = (const float*)d_in[18];
  const float* bh1          = (const float*)d_in[19];
  const float* Wh2          = (const float*)d_in[20];
  const float* bh2          = (const float*)d_in[21];
  float* out = (float*)d_out;

  // workspace layout (bytes)
  char* ws = (char*)d_ws;
  size_t off = 0;
  auto take = [&](size_t bytes) { char* p = ws + off; off += (bytes + 255) & ~(size_t)255; return p; };
  f16* Xh       = (f16*)take((size_t)32 * 1024 * 1024 * 2);   // 64 MB
  f16* Wkh      = (f16*)take((size_t)1024 * 1024 * 2);
  f16* Wvh      = (f16*)take((size_t)1024 * 1024 * 2);
  f16* Wqh      = (f16*)take((size_t)1024 * 1024 * 2);
  f16* Woh      = (f16*)take((size_t)1024 * 1024 * 2);
  f16* W1h      = (f16*)take((size_t)512 * 2048 * 2);
  f16* Kh       = (f16*)take((size_t)32 * 1024 * 1024 * 2);   // 64 MB
  f16* Vh       = (f16*)take((size_t)32 * 1024 * 1024 * 2);   // 64 MB
  f16* queryh   = (f16*)take((size_t)480 * 1024 * 2);
  f16* qrow     = (f16*)take((size_t)480 * 1024 * 2);
  f16* qh       = (f16*)take((size_t)32 * 8 * 16 * 128 * 2);
  f16* ctxh     = (f16*)take((size_t)480 * 1024 * 2);
  float* enrich = (float*)take((size_t)480 * 1024 * 4);
  f16* enrichh  = (f16*)take((size_t)480 * 1024 * 2);
  float* h_i    = (float*)take((size_t)480 * 512 * 4);
  float* h_j    = (float*)take((size_t)480 * 512 * 4);

  // stage f16 operands
  cast_vf<<<131072, 256, 0, stream>>>(visual_feats, Xh);
  cvt_f32_f16<<<4096, 256, 0, stream>>>(Wk, Wkh, 1024 * 1024);
  cvt_f32_f16<<<4096, 256, 0, stream>>>(Wv, Wvh, 1024 * 1024);
  cvt_f32_f16<<<4096, 256, 0, stream>>>(Wq, Wqh, 1024 * 1024);
  cvt_f32_f16<<<4096, 256, 0, stream>>>(Wo, Woh, 1024 * 1024);
  cvt_f32_f16<<<4096, 256, 0, stream>>>(W1, W1h, 512 * 2048);

  // query path
  query_kernel<<<1920, 256, 0, stream>>>(nodes, bboxes, emb, Wb, bb, queryh);
  gemm16_wmma<<<dim3(64, 30), 32, 0, stream>>>(queryh, 1024, Wqh, 1024, bq,
                                               nullptr, qrow, 1024, 1024);
  repack_q<<<2048, 256, 0, stream>>>(qrow, qh);

  // K/V projections: LDS-tiled WMMA GEMM, tiles filled by the Tensor Data Mover
  gemm_tdm_wmma<<<dim3(8, 256), 256, 0, stream>>>(Xh, Wkh, bk, Kh);
  gemm_tdm_wmma<<<dim3(8, 256), 256, 0, stream>>>(Xh, Wvh, bv, Vh);

  // attention
  attn_kernel<<<dim3(8, 32), 256, 0, stream>>>(qh, Kh, Vh, ctxh);

  // enriched = ctx @ Wo^T + bo  (need f32 for energy, f16 for next GEMMs)
  gemm16_wmma<<<dim3(64, 30), 32, 0, stream>>>(ctxh, 1024, Woh, 1024, bo,
                                               enrich, enrichh, 1024, 1024);

  // h_i / h_j = enriched @ W1[:, :D]^T / W1[:, D:]^T  (ldw = 2048)
  gemm16_wmma<<<dim3(32, 30), 32, 0, stream>>>(enrichh, 1024, W1h, 2048, nullptr,
                                               h_i, nullptr, 512, 1024);
  gemm16_wmma<<<dim3(32, 30), 32, 0, stream>>>(enrichh, 1024, W1h + 1024, 2048, nullptr,
                                               h_j, nullptr, 512, 1024);

  // edge logits -> d_out[0 .. 129600)
  edge_kernel<<<507, 256, 0, stream>>>(h_i, h_j, b1, W2, b2, out);

  // energy -> d_out[129600 .. 129632)
  energy_kernel<<<32, 256, 0, stream>>>(enrich, nodes, Wh1, bh1, Wh2, bh2,
                                        out + 129600);
}
